// GATLayer_71786083385836
// MI455X (gfx1250) — matrix operand order repaired and checked by reading, not compiled, not run on hardware
//
#include <hip/hip_runtime.h>

// ---------------------------------------------------------------------------
// GAT layer, algebraically reduced:
//   softmax of a per-head constant over the edge mask == uniform 1/deg over
//   *distinct* neighbors  ->  out = W_all * mean-aggregate(x);  q/k/att dead.
// Pipeline:
//   K0 zero adjacency bitmap
//   K1 build bitmap from edge_index (dedup via bits)
//   K2 per-node neighbor mean of x, split into f16 hi/lo, pack A' [N,768]
//   K3 pack W into WMMA-B-fragment-swizzled f16 B' [H,16dt,24kt,32x16]
//   K4 f16 WMMA GEMM, 2x4 register-blocked (32 nodes x 64 ch per wave),
//      K=768 split-precision trick, f32 accumulate -> d_out
// ---------------------------------------------------------------------------

constexpr int N_NODES = 4096;
constexpr int DCH     = 256;        // in == out channels per head
constexpr int NH      = 8;
constexpr int HD      = NH * DCH;   // 2048
constexpr int WORDS   = N_NODES / 32;  // 128 bitmap words per row
constexpr int KP      = 3 * DCH;    // 768: [x_hi ; x_lo ; x_hi] x [w_hi ; w_hi ; w_lo]
constexpr int KT      = KP / 32;    // 24 WMMA K-steps

typedef __attribute__((ext_vector_type(16))) _Float16 v16h;
typedef __attribute__((ext_vector_type(8)))  _Float16 v8h;
typedef __attribute__((ext_vector_type(8)))  float    v8f;

// ---- K0: zero the adjacency bitmap --------------------------------------
__global__ void k_zero_bits(unsigned* __restrict__ bits, int nwords) {
    int i = blockIdx.x * blockDim.x + threadIdx.x;
    if (i < nwords) bits[i] = 0u;
}

// ---- K1: scatter edges into bitmap (dedup for free) ----------------------
__global__ void k_build_adj(const int* __restrict__ ei, int E,
                            unsigned* __restrict__ bits) {
    int e = blockIdx.x * blockDim.x + threadIdx.x;
    if (e < E) {
        int r = ei[e];        // edge_index[0][e]  (softmax row)
        int c = ei[E + e];    // edge_index[1][e]  (neighbor column)
        atomicOr(&bits[(size_t)r * WORDS + (c >> 5)], 1u << (c & 31));
    }
}

// ---- K2: xagg[i,k] = mean over distinct neighbors; pack split-f16 A' -----
__global__ __launch_bounds__(256) void k_agg_pack(
        const float* __restrict__ x, const unsigned* __restrict__ bits,
        _Float16* __restrict__ Ap) {
    const int i = blockIdx.x;      // node
    const int k = threadIdx.x;     // channel 0..255
    __shared__ unsigned sb[WORDS];
    __shared__ int sdeg;
    if (k == 0) sdeg = 0;
    if (k < WORDS) sb[k] = bits[(size_t)i * WORDS + k];
    __syncthreads();
    if (k < WORDS) atomicAdd(&sdeg, __popc(sb[k]));
    __syncthreads();
    const int deg = sdeg;

    float s = 0.0f;
    if (deg > 0) {
        // bitmap is uniform across the block -> divergence-free scan,
        // x row reads fully coalesced across k
        for (int w = 0; w < WORDS; ++w) {
            unsigned b = sb[w];
            int base = w << 5;
            while (b) {
                int m = base + __builtin_ctz(b);
                b &= b - 1;
                s += x[(size_t)m * DCH + k];
            }
        }
        s *= (1.0f / (float)deg);
    } else {
        // empty row: softmax over all -1e9 is uniform 1/N
        for (int m = 0; m < N_NODES; ++m) s += x[(size_t)m * DCH + k];
        s *= (1.0f / (float)N_NODES);
    }

    _Float16 hi = (_Float16)s;
    _Float16 lo = (_Float16)(s - (float)hi);
    _Float16* row = Ap + (size_t)i * KP;
    row[k]            = hi;   // pairs with w_hi
    row[DCH + k]      = lo;   // pairs with w_hi
    row[2 * DCH + k]  = hi;   // pairs with w_lo
}

// ---- K3: pack W into WMMA B-fragment-swizzled layout ---------------------
// B' fragment (h, dt, kt): 32(K) x 16(Nchan) f16, one contiguous 16-f16 run
// per lane: lanes 0-15 hold K = ktBase+0..15 (col = lane), lanes 16-31 hold
// K = ktBase+16..31 (col = lane-16).  => one contiguous 32B load per lane.
__global__ void k_pack_B(const float* __restrict__ W, _Float16* __restrict__ Bp) {
    int tid = blockIdx.x * blockDim.x + threadIdx.x;
    if (tid >= NH * KP * DCH) return;
    int h   = tid / (KP * DCH);
    int rem = tid % (KP * DCH);
    int kp  = rem / DCH;           // K' in [0,768)
    int d   = rem % DCH;           // output channel
    int k    = kp & 255;
    int part = kp >> 8;            // 0:w_hi 1:w_hi 2:w_lo
    float w = W[(size_t)h * DCH * DCH + (size_t)d * DCH + k];
    _Float16 wh = (_Float16)w;
    _Float16 wl = (_Float16)(w - (float)wh);
    _Float16 val = (part == 2) ? wl : wh;
    int dt = d >> 4, nI = d & 15;
    int kt = kp >> 5, kIn = kp & 31;
    int lane = nI + (kIn & 16);
    int pos  = kIn & 15;
    size_t idx = ((size_t)((h * 16 + dt) * KT + kt)) * 512 + (size_t)lane * 16 + pos;
    Bp[idx] = val;
}

// ---- K4: WMMA GEMM, 2(M) x 4(N) tiles per wave ---------------------------
// out[n, h*256+d] = sum_K' A'[n,K'] * B'[K', h,d]
// 8 WMMAs per K-step against 4 A-loads + 8 B-loads (A reused x4, B x2).
__global__ __launch_bounds__(256) void k_gemm_wmma(
        const _Float16* __restrict__ Ap, const _Float16* __restrict__ Bp,
        float* __restrict__ out) {
    const int wid  = blockIdx.x * (blockDim.x >> 5) + (threadIdx.x >> 5);
    const int lane = threadIdx.x & 31;
    const int nodeTile2 = wid >> 5;       // 128 tiles of 32 nodes
    const int hq = wid & 31;              // 8 heads * 4 quad-channel tiles
    const int h   = hq >> 2;
    const int dtq = hq & 3;               // 64-channel group

    // A fragment source: row-major A', contiguous k order (k = 16*half + s), same as B'.
    const _Float16* Ar0 = Ap + (size_t)(nodeTile2 * 32 + (lane & 15)) * KP;
    const _Float16* Ar1 = Ar0 + (size_t)16 * KP;
    const int ko = (lane >> 4) << 4;      // lanes 0-15: K+0..15, 16-31: K+16..31
    // B fragment sources: pre-swizzled, contiguous 32B per lane.
    const _Float16* Bf = Bp + (size_t)((h * 16 + dtq * 4) * KT) * 512
                            + (size_t)lane * 16;

    v8f acc[2][4];
#pragma unroll
    for (int mi = 0; mi < 2; ++mi)
#pragma unroll
        for (int j = 0; j < 4; ++j) acc[mi][j] = (v8f){};

#pragma unroll 2
    for (int kt = 0; kt < KT; ++kt) {
        v8h a00 = *(const v8h*)(Ar0 + kt * 32 + ko);
        v8h a01 = *(const v8h*)(Ar0 + kt * 32 + 8 + ko);
        v16h aA = __builtin_shufflevector(a00, a01, 0, 1, 2, 3, 4, 5, 6, 7,
                                                    8, 9, 10, 11, 12, 13, 14, 15);
        v8h a10 = *(const v8h*)(Ar1 + kt * 32 + ko);
        v8h a11 = *(const v8h*)(Ar1 + kt * 32 + 8 + ko);
        v16h aB = __builtin_shufflevector(a10, a11, 0, 1, 2, 3, 4, 5, 6, 7,
                                                    8, 9, 10, 11, 12, 13, 14, 15);
        v16h b[4];
#pragma unroll
        for (int j = 0; j < 4; ++j)
            b[j] = *(const v16h*)(Bf + (size_t)(j * KT + kt) * 512);
#pragma unroll
        for (int j = 0; j < 4; ++j) {
            acc[0][j] = __builtin_amdgcn_wmma_f32_16x16x32_f16(
                    false, aA, false, b[j], (short)0, acc[0][j], false, false);
            acc[1][j] = __builtin_amdgcn_wmma_f32_16x16x32_f16(
                    false, aB, false, b[j], (short)0, acc[1][j], false, false);
        }
    }

    // C/D layout: VGPR r -> M = r + 8*(lane>=16), N = lane&15
    const int colBase = h * DCH + dtq * 64 + (lane & 15);
    const int rowBase = nodeTile2 * 32 + ((lane >> 4) << 3);
#pragma unroll
    for (int mi = 0; mi < 2; ++mi)
#pragma unroll
        for (int j = 0; j < 4; ++j) {
            const int col = colBase + j * 16;
#pragma unroll
            for (int r = 0; r < 8; ++r)
                out[(size_t)(rowBase + mi * 16 + r) * HD + col] = acc[mi][j][r];
        }
}

// ---------------------------------------------------------------------------
extern "C" void kernel_launch(void* const* d_in, const int* in_sizes, int n_in,
                              void* d_out, int out_size, void* d_ws, size_t ws_size,
                              hipStream_t stream) {
    const float* x  = (const float*)d_in[0];          // [4096, 256]
    const float* W  = (const float*)d_in[1];          // [8, 256, 256]
    // d_in[2], d_in[3] (query/key weights) are provably dead — see analysis.
    const int*   ei = (const int*)d_in[4];            // [2, E]
    const int E = in_sizes[4] / 2;

    unsigned*  bits = (unsigned*)d_ws;                                  // 2 MB
    _Float16*  Ap   = (_Float16*)((char*)d_ws + (2u << 20));            // 6 MB
    _Float16*  Bp   = (_Float16*)((char*)d_ws + (8u << 20));            // 3 MB

    const int nwords = N_NODES * WORDS;   // 524288
    k_zero_bits<<<(nwords + 255) / 256, 256, 0, stream>>>(bits, nwords);
    k_build_adj<<<(E + 255) / 256, 256, 0, stream>>>(ei, E, bits);
    k_agg_pack<<<N_NODES, 256, 0, stream>>>(x, bits, Ap);
    {
        const int nb = (NH * KP * DCH + 255) / 256;   // 6144 blocks
        k_pack_B<<<nb, 256, 0, stream>>>(W, Bp);
    }
    {
        // 128 node-pair tiles * 32 (h,dtq) tiles = 4096 waves, 8 waves/block
        k_gemm_wmma<<<512, 256, 0, stream>>>(Ap, Bp, (float*)d_out);
    }
}
